// AttentionRGCN_56066503082345
// MI455X (gfx1250) — matrix operand order repaired and checked by reading, not compiled, not run on hardware
//
#include <hip/hip_runtime.h>

#define N_NODES 50000
#define N_PAD   50176   // 196 * 256 : node rows padded so every GEMM wave is full
#define N_EDGES 600000
#define RELS 8
#define BASES 8
#define DIM 128
#define HEADS 4
#define HD (HEADS * DIM)
#define NEG_SLOPE 0.1f
#define LN_EPS 1e-5f

typedef __bf16 bf16;
typedef __attribute__((ext_vector_type(16))) __bf16 v16bf;
typedef __attribute__((ext_vector_type(8)))  float  v8f;

// ---------- small helpers ----------
__device__ inline float wave_sum32(float v) {
#pragma unroll
    for (int o = 16; o > 0; o >>= 1) v += __shfl_xor(v, o, 32);
    return v;
}
__device__ inline unsigned f2ord(float f) {
    unsigned u = __float_as_uint(f);
    return (u & 0x80000000u) ? ~u : (u | 0x80000000u);
}
__device__ inline float ord2f(unsigned u) {
    return __uint_as_float((u & 0x80000000u) ? (u & 0x7fffffffu) : ~u);
}
__device__ inline float leaky(float x, float s) { return x > 0.0f ? x : s * x; }

// ---------- generic utility kernels ----------
__global__ void zero_f32_kernel(float* __restrict__ p, int n) {
    int i = blockIdx.x * blockDim.x + threadIdx.x;
    if (i < n) p[i] = 0.0f;
}

// copy n_valid elements, zero-fill to n_total (pads node features)
__global__ void copy_pad_kernel(const float* __restrict__ in, float* __restrict__ out,
                                int n_valid, int n_total) {
    int i = blockIdx.x * blockDim.x + threadIdx.x;
    if (i < n_total) out[i] = (i < n_valid) ? in[i] : 0.0f;
}

// in[rows][cols] (f32) -> out[cols][rows] (bf16)
__global__ void transpose_bf16_kernel(const float* __restrict__ in, bf16* __restrict__ out,
                                      int rows, int cols) {
    int idx = blockIdx.x * blockDim.x + threadIdx.x;
    if (idx >= rows * cols) return;
    int c = idx / rows;
    int r = idx - c * rows;
    out[idx] = (bf16)in[(size_t)r * cols + c];
}

// Wt[r][o][i] = sum_b comp[r,b] * basis[b,i,o]   (pre-transposed bf16 weights)
__global__ void prep_relW_kernel(const float* __restrict__ basis, const float* __restrict__ comp,
                                 bf16* __restrict__ Wt) {
    int idx = blockIdx.x * blockDim.x + threadIdx.x;
    if (idx >= RELS * DIM * DIM) return;
    int r   = idx / (DIM * DIM);
    int rem = idx - r * DIM * DIM;
    int o   = rem / DIM;
    int i   = rem - o * DIM;
    float s = 0.0f;
#pragma unroll
    for (int b = 0; b < BASES; ++b)
        s += comp[r * BASES + b] * basis[((size_t)b * DIM + i) * DIM + o];
    Wt[idx] = (bf16)s;
}

// deg[et*N_PAD + dst] += 1   (padded per-relation stride)
__global__ void deg_kernel(const int* __restrict__ dst, const int* __restrict__ et,
                           float* __restrict__ deg) {
    int e = blockIdx.x * blockDim.x + threadIdx.x;
    if (e >= N_EDGES) return;
    atomicAdd(&deg[(size_t)et[e] * N_PAD + dst[e]], 1.0f);
}
__global__ void inv_kernel(float* __restrict__ d) {
    int i = blockIdx.x * blockDim.x + threadIdx.x;
    if (i < RELS * N_PAD) d[i] = 1.0f / fmaxf(d[i], 1.0f);
}

// ---------- WMMA bf16 GEMM: C[M x NOUT] = (scale?)A[M x 128] @ Bt[NOUT x 128]^T ----------
// REQUIREMENT: M % 256 == 0 and grid.x == M/256 (we always pass M = N_PAD).
// 32 rows per wave (two 16-row accumulator groups sharing each B fragment),
// 8 waves per block -> 256 rows per block. Completely guard-free.
template <int NOUT, bool ACCUM, bool ADD_BIAS, bool ROWSCALE>
__global__ __launch_bounds__(256) void gemm_bf16_kernel(const float* __restrict__ A,
                                                        const bf16* __restrict__ Bt,
                                                        const float* __restrict__ rowscale,
                                                        const float* __restrict__ bias,
                                                        float* __restrict__ C) {
    const int lane = threadIdx.x & 31;
    const int wave = threadIdx.x >> 5;
    const int row0 = (blockIdx.x * 8 + wave) * 32;
    const int m = lane & 15;  // row within A-frag / col within B-frag
    const int h = lane >> 4;  // lane half

    // Two A-fragment sets (rows row0..+15 and row0+16..+31), all 4 k-steps.
    // A layout (matches B): lane-half h holds K=[16h,16h+16) contiguous per 32-K step.
    v16bf afrag[2][4];
#pragma unroll
    for (int g = 0; g < 2; ++g) {
        const int arow = row0 + g * 16 + m;
        const float scale = ROWSCALE ? rowscale[arow] : 1.0f;
        const float* Arow = A + (size_t)arow * DIM;
#pragma unroll
        for (int kk = 0; kk < 4; ++kk) {
            const float* p0 = Arow + kk * 32 + h * 16;
            const float* p1 = Arow + kk * 32 + h * 16 + 8;
            v16bf a;
#pragma unroll
            for (int j = 0; j < 8; ++j) {
                a[j]     = (bf16)(p0[j] * scale);
                a[8 + j] = (bf16)(p1[j] * scale);
            }
            afrag[g][kk] = a;
        }
    }

    for (int t = 0; t < NOUT / 16; ++t) {
        const int col0 = t * 16;

        // Preload all 4 B fragments (one clause), then 8 back-to-back WMMAs.
        // B layout: lane holds column n=lane%16, K = 16*(lane/16) .. +15 contiguous.
        const bf16* Brow = Bt + (size_t)(col0 + m) * DIM;
        v16bf bfrag[4];
#pragma unroll
        for (int kk = 0; kk < 4; ++kk) bfrag[kk] = *(const v16bf*)(Brow + kk * 32 + h * 16);

        v8f acc[2];
#pragma unroll
        for (int g = 0; g < 2; ++g) {
            const int rbase = row0 + g * 16 + 8 * h;
            if (ACCUM) {
                const float* Cp = C + (size_t)rbase * NOUT + col0 + m;
#pragma unroll
                for (int v = 0; v < 8; ++v) acc[g][v] = Cp[(size_t)v * NOUT];
            } else if (ADD_BIAS) {
                float bv = bias[col0 + m];
#pragma unroll
                for (int v = 0; v < 8; ++v) acc[g][v] = bv;
            } else {
#pragma unroll
                for (int v = 0; v < 8; ++v) acc[g][v] = 0.0f;
            }
        }

#pragma unroll
        for (int kk = 0; kk < 4; ++kk) {
            acc[0] = __builtin_amdgcn_wmma_f32_16x16x32_bf16(false, afrag[0][kk], false, bfrag[kk],
                                                             (short)0, acc[0], false, false);
            acc[1] = __builtin_amdgcn_wmma_f32_16x16x32_bf16(false, afrag[1][kk], false, bfrag[kk],
                                                             (short)0, acc[1], false, false);
        }

#pragma unroll
        for (int g = 0; g < 2; ++g) {
            float* Cp = C + (size_t)(row0 + g * 16 + 8 * h) * NOUT + col0 + m;
#pragma unroll
            for (int v = 0; v < 8; ++v) Cp[(size_t)v * NOUT] = acc[g][v];
        }
    }
}

// ---------- RGCN edge scatter (relation at a time), wave per edge ----------
__global__ __launch_bounds__(256) void scatter_rel_kernel(const int* __restrict__ src,
                                                          const int* __restrict__ dst,
                                                          const int* __restrict__ et,
                                                          const float* __restrict__ x,
                                                          float* __restrict__ agg, int rel) {
    int e = blockIdx.x * 8 + (threadIdx.x >> 5);
    if (e >= N_EDGES) return;
    if (et[e] != rel) return;
    int lane = threadIdx.x & 31;
    const float* xs = x + (size_t)src[e] * DIM;
    float* ad = agg + (size_t)dst[e] * DIM;
#pragma unroll
    for (int j = 0; j < 4; ++j) atomicAdd(&ad[lane + 32 * j], xs[lane + 32 * j]);
}

// ---------- LayerNorm + leaky_relu, wave per node ----------
__global__ __launch_bounds__(256) void ln_leaky_kernel(const float* __restrict__ in,
                                                       float* __restrict__ out,
                                                       const float* __restrict__ g,
                                                       const float* __restrict__ b) {
    int n = blockIdx.x * 8 + (threadIdx.x >> 5);
    if (n >= N_NODES) return;
    int lane = threadIdx.x & 31;
    const float* xr = in + (size_t)n * DIM;
    float v[4];
    float s = 0.0f;
#pragma unroll
    for (int j = 0; j < 4; ++j) { v[j] = xr[lane + 32 * j]; s += v[j]; }
    float mu = wave_sum32(s) * (1.0f / DIM);
    float vs = 0.0f;
#pragma unroll
    for (int j = 0; j < 4; ++j) { float d = v[j] - mu; vs += d * d; }
    float var = wave_sum32(vs) * (1.0f / DIM);
    float rs = rsqrtf(var + LN_EPS);
#pragma unroll
    for (int j = 0; j < 4; ++j) {
        int d = lane + 32 * j;
        float y = (v[j] - mu) * rs * g[d] + b[d];
        out[(size_t)n * DIM + d] = leaky(y, NEG_SLOPE);
    }
}

// ---------- GAT ----------
__global__ __launch_bounds__(256) void gat_scores_kernel(const float* __restrict__ hH,
                                                         const float* __restrict__ asrc,
                                                         const float* __restrict__ adst,
                                                         float* __restrict__ ssrc,
                                                         float* __restrict__ sdst,
                                                         unsigned* __restrict__ mbuf) {
    int n = blockIdx.x * 8 + (threadIdx.x >> 5);
    if (n >= N_NODES) return;
    int lane = threadIdx.x & 31;
#pragma unroll
    for (int hh = 0; hh < HEADS; ++hh) {
        const float* hp = hH + (size_t)n * HD + hh * DIM;
        float s1 = 0.0f, s2 = 0.0f;
#pragma unroll
        for (int j = 0; j < 4; ++j) {
            float hv = hp[lane + 32 * j];
            s1 += hv * asrc[hh * DIM + lane + 32 * j];
            s2 += hv * adst[hh * DIM + lane + 32 * j];
        }
        s1 = wave_sum32(s1);
        s2 = wave_sum32(s2);
        if (lane == 0) {
            ssrc[n * HEADS + hh] = s1;
            sdst[n * HEADS + hh] = s2;
            mbuf[n * HEADS + hh] = f2ord(leaky(s1 + s2, 0.2f));  // self-loop
        }
    }
}

__global__ void gat_edge_max_kernel(const int* __restrict__ src, const int* __restrict__ dst,
                                    const float* __restrict__ ssrc, const float* __restrict__ sdst,
                                    unsigned* __restrict__ mbuf) {
    int e = blockIdx.x * blockDim.x + threadIdx.x;
    if (e >= N_EDGES) return;
    int s = src[e], d = dst[e];
#pragma unroll
    for (int hh = 0; hh < HEADS; ++hh) {
        float ev = leaky(ssrc[s * HEADS + hh] + sdst[d * HEADS + hh], 0.2f);
        atomicMax(&mbuf[d * HEADS + hh], f2ord(ev));
    }
}

__global__ void gat_denom_init_kernel(const float* __restrict__ ssrc,
                                      const float* __restrict__ sdst,
                                      const unsigned* __restrict__ mbuf,
                                      float* __restrict__ denom) {
    int i = blockIdx.x * blockDim.x + threadIdx.x;
    if (i >= N_NODES * HEADS) return;
    float e = leaky(ssrc[i] + sdst[i], 0.2f);
    denom[i] = expf(e - ord2f(mbuf[i]));
}

__global__ void gat_denom_edge_kernel(const int* __restrict__ src, const int* __restrict__ dst,
                                      const float* __restrict__ ssrc,
                                      const float* __restrict__ sdst,
                                      const unsigned* __restrict__ mbuf,
                                      float* __restrict__ denom) {
    int e = blockIdx.x * blockDim.x + threadIdx.x;
    if (e >= N_EDGES) return;
    int s = src[e], d = dst[e];
#pragma unroll
    for (int hh = 0; hh < HEADS; ++hh) {
        int idx = d * HEADS + hh;
        float ev = leaky(ssrc[s * HEADS + hh] + sdst[idx], 0.2f);
        atomicAdd(&denom[idx], expf(ev - ord2f(mbuf[idx])));
    }
}

// out[n,d] = gat_bias[d] + (1/H) * sum_h coef_self * hH[n,h,d]
__global__ void gat_agg_init_kernel(const float* __restrict__ hH, const float* __restrict__ ssrc,
                                    const float* __restrict__ sdst,
                                    const unsigned* __restrict__ mbuf,
                                    const float* __restrict__ denom,
                                    const float* __restrict__ bias, float* __restrict__ out) {
    int i = blockIdx.x * blockDim.x + threadIdx.x;
    if (i >= N_NODES * DIM) return;
    int n = i / DIM, d = i - n * DIM;
    float acc = bias[d];
#pragma unroll
    for (int hh = 0; hh < HEADS; ++hh) {
        int idx = n * HEADS + hh;
        float ev = leaky(ssrc[idx] + sdst[idx], 0.2f);
        float coef = expf(ev - ord2f(mbuf[idx])) / denom[idx];
        acc += 0.25f * coef * hH[(size_t)n * HD + hh * DIM + d];
    }
    out[(size_t)n * DIM + d] = acc;
}

// wave per edge: out[dst,d] += (1/H) sum_h coef[h] * hH[src,h,d]
__global__ __launch_bounds__(256) void gat_agg_edge_kernel(const int* __restrict__ src,
                                                           const int* __restrict__ dst,
                                                           const float* __restrict__ hH,
                                                           const float* __restrict__ ssrc,
                                                           const float* __restrict__ sdst,
                                                           const unsigned* __restrict__ mbuf,
                                                           const float* __restrict__ denom,
                                                           float* __restrict__ out) {
    int e = blockIdx.x * 8 + (threadIdx.x >> 5);
    if (e >= N_EDGES) return;
    int lane = threadIdx.x & 31;
    int s = src[e], d = dst[e];
    float coef[HEADS];
#pragma unroll
    for (int hh = 0; hh < HEADS; ++hh) {
        int idx = d * HEADS + hh;
        float ev = leaky(ssrc[s * HEADS + hh] + sdst[idx], 0.2f);
        coef[hh] = 0.25f * expf(ev - ord2f(mbuf[idx])) / denom[idx];
    }
    const float* hp = hH + (size_t)s * HD;
    float* op = out + (size_t)d * DIM;
#pragma unroll
    for (int j = 0; j < 4; ++j) {
        int dd = lane + 32 * j;
        float v = 0.0f;
#pragma unroll
        for (int hh = 0; hh < HEADS; ++hh) v += coef[hh] * hp[hh * DIM + dd];
        atomicAdd(&op[dd], v);
    }
}

// ---------- final row L2 normalize, wave per node ----------
__global__ __launch_bounds__(256) void l2norm_kernel(const float* __restrict__ in,
                                                     float* __restrict__ out) {
    int n = blockIdx.x * 8 + (threadIdx.x >> 5);
    if (n >= N_NODES) return;
    int lane = threadIdx.x & 31;
    const float* xr = in + (size_t)n * DIM;
    float v[4];
    float ss = 0.0f;
#pragma unroll
    for (int j = 0; j < 4; ++j) { v[j] = xr[lane + 32 * j]; ss += v[j] * v[j]; }
    float nrm = sqrtf(wave_sum32(ss));
    float inv = 1.0f / fmaxf(nrm, 1e-12f);
#pragma unroll
    for (int j = 0; j < 4; ++j) out[(size_t)n * DIM + lane + 32 * j] = v[j] * inv;
}

// ---------- host side ----------
extern "C" void kernel_launch(void* const* d_in, const int* in_sizes, int n_in,
                              void* d_out, int out_size, void* d_ws, size_t ws_size,
                              hipStream_t stream) {
    const float* x       = (const float*)d_in[0];
    const int* ei        = (const int*)d_in[1];
    const int* src       = ei;
    const int* dst       = ei + N_EDGES;
    const int* et        = (const int*)d_in[2];
    const float* r_basis[3] = {(const float*)d_in[3], (const float*)d_in[9], (const float*)d_in[21]};
    const float* r_comp[3]  = {(const float*)d_in[4], (const float*)d_in[10], (const float*)d_in[22]};
    const float* r_root[3]  = {(const float*)d_in[5], (const float*)d_in[11], (const float*)d_in[23]};
    const float* r_bias[3]  = {(const float*)d_in[6], (const float*)d_in[12], (const float*)d_in[24]};
    const float* ln_g[3]    = {(const float*)d_in[7], (const float*)d_in[13], (const float*)d_in[19]};
    const float* ln_b[3]    = {(const float*)d_in[8], (const float*)d_in[14], (const float*)d_in[20]};
    const float* gat_w    = (const float*)d_in[15];
    const float* gat_asrc = (const float*)d_in[16];
    const float* gat_adst = (const float*)d_in[17];
    const float* gat_bias = (const float*)d_in[18];

    // workspace carve-up (256B aligned); node-major buffers padded to N_PAD rows
    size_t off = 0;
    char* base = (char*)d_ws;
    auto carve = [&](size_t bytes) {
        void* p = base + off;
        off += (bytes + 255) & ~(size_t)255;
        return p;
    };
    float* featA   = (float*)carve((size_t)N_PAD * DIM * 4);
    float* featB   = (float*)carve((size_t)N_PAD * DIM * 4);
    float* agg     = (float*)carve((size_t)N_PAD * DIM * 4);
    float* hH      = (float*)carve((size_t)N_PAD * HD * 4);
    float* invbuf  = (float*)carve((size_t)RELS * N_PAD * 4);
    bf16* Wt       = (bf16*)carve((size_t)RELS * DIM * DIM * 2);
    bf16* rootT    = (bf16*)carve((size_t)DIM * DIM * 2);
    bf16* gatWt    = (bf16*)carve((size_t)DIM * HD * 2);
    float* ssrc    = (float*)carve((size_t)N_NODES * HEADS * 4);
    float* sdst    = (float*)carve((size_t)N_NODES * HEADS * 4);
    unsigned* mbuf = (unsigned*)carve((size_t)N_NODES * HEADS * 4);
    float* denom   = (float*)carve((size_t)N_NODES * HEADS * 4);
    (void)ws_size; (void)n_in; (void)in_sizes; (void)out_size;

    const int TPB = 256;
    const int gemm_grid = N_PAD / 256;                // 8 waves * 32 rows per block
    const int wave_edge_grid = (N_EDGES + 7) / 8;     // wave per edge
    const int wave_node_grid = (N_NODES + 7) / 8;     // wave per node

    // stage input into padded featA (zero pad rows)
    copy_pad_kernel<<<(N_PAD * DIM + TPB - 1) / TPB, TPB, 0, stream>>>(
        x, featA, N_NODES * DIM, N_PAD * DIM);

    // degree -> inverse (padded per-relation stride; shared by all RGCN layers)
    zero_f32_kernel<<<(RELS * N_PAD + TPB - 1) / TPB, TPB, 0, stream>>>(invbuf, RELS * N_PAD);
    deg_kernel<<<(N_EDGES + TPB - 1) / TPB, TPB, 0, stream>>>(dst, et, invbuf);
    inv_kernel<<<(RELS * N_PAD + TPB - 1) / TPB, TPB, 0, stream>>>(invbuf);

    auto rgcn_layer = [&](const float* xin, float* xout, int L) {
        prep_relW_kernel<<<(RELS * DIM * DIM + TPB - 1) / TPB, TPB, 0, stream>>>(
            r_basis[L], r_comp[L], Wt);
        transpose_bf16_kernel<<<(DIM * DIM + TPB - 1) / TPB, TPB, 0, stream>>>(
            r_root[L], rootT, DIM, DIM);
        // xout = xin @ root + bias
        gemm_bf16_kernel<DIM, false, true, false><<<gemm_grid, TPB, 0, stream>>>(
            xin, rootT, nullptr, r_bias[L], xout);
        for (int r = 0; r < RELS; ++r) {
            zero_f32_kernel<<<(N_PAD * DIM + TPB - 1) / TPB, TPB, 0, stream>>>(
                agg, N_PAD * DIM);
            scatter_rel_kernel<<<wave_edge_grid, TPB, 0, stream>>>(src, dst, et, xin, agg, r);
            // xout += (inv[r] ⊙ agg) @ W[r]
            gemm_bf16_kernel<DIM, true, false, true><<<gemm_grid, TPB, 0, stream>>>(
                agg, Wt + (size_t)r * DIM * DIM, invbuf + (size_t)r * N_PAD, nullptr, xout);
        }
    };

    // layer 0 (input staged in featA)
    rgcn_layer(featA, featB, 0);
    ln_leaky_kernel<<<wave_node_grid, TPB, 0, stream>>>(featB, featA, ln_g[0], ln_b[0]);
    // layer 1
    rgcn_layer(featA, featB, 1);
    ln_leaky_kernel<<<wave_node_grid, TPB, 0, stream>>>(featB, featA, ln_g[1], ln_b[1]);

    // GAT
    transpose_bf16_kernel<<<(DIM * HD + TPB - 1) / TPB, TPB, 0, stream>>>(gat_w, gatWt, DIM, HD);
    gemm_bf16_kernel<HD, false, false, false><<<gemm_grid, TPB, 0, stream>>>(
        featA, gatWt, nullptr, nullptr, hH);
    gat_scores_kernel<<<wave_node_grid, TPB, 0, stream>>>(hH, gat_asrc, gat_adst, ssrc, sdst, mbuf);
    gat_edge_max_kernel<<<(N_EDGES + TPB - 1) / TPB, TPB, 0, stream>>>(src, dst, ssrc, sdst, mbuf);
    gat_denom_init_kernel<<<(N_NODES * HEADS + TPB - 1) / TPB, TPB, 0, stream>>>(
        ssrc, sdst, mbuf, denom);
    gat_denom_edge_kernel<<<(N_EDGES + TPB - 1) / TPB, TPB, 0, stream>>>(
        src, dst, ssrc, sdst, mbuf, denom);
    gat_agg_init_kernel<<<(N_NODES * DIM + TPB - 1) / TPB, TPB, 0, stream>>>(
        hH, ssrc, sdst, mbuf, denom, gat_bias, featB);
    gat_agg_edge_kernel<<<wave_edge_grid, TPB, 0, stream>>>(
        src, dst, hH, ssrc, sdst, mbuf, denom, featB);
    ln_leaky_kernel<<<wave_node_grid, TPB, 0, stream>>>(featB, featA, ln_g[2], ln_b[2]);

    // layer 2 + final normalize
    rgcn_layer(featA, featB, 2);
    l2norm_kernel<<<wave_node_grid, TPB, 0, stream>>>(featB, (float*)d_out);
}
